// WideAndDeepModel_40200893890818
// MI455X (gfx1250) — matrix operand order repaired and checked by reading, not compiled, not run on hardware
//
#include <hip/hip_runtime.h>
#include <stdint.h>

// Problem constants (from reference)
#define NF 3
#define NC 256
#define ED 64
#define NH 1024
#define NDN 13
#define NB 16384
#define DEEP_IN 205    // NF*ED + NDN
#define DEEP_K  224    // padded to multiple of 32 for bf16 WMMA

// wide_w offsets (replicating reference, including the cross = s*F (+ not *C) quirk)
#define OFF_P01   768        // NF*NC
#define OFF_P02   66304      // 768 + 65536
#define OFF_P12   131840     // 768 + 2*65536
#define OFF_T012  197376     // 768 + 3*65536
#define OFF_DENSE 16974592   // 197376 + 256^3

typedef __attribute__((ext_vector_type(16))) __bf16 v16bf;
typedef __attribute__((ext_vector_type(8)))  float  v8f;

union Frag { v16bf v; uint4 q[2]; };

__device__ __forceinline__ unsigned short f2bf(float f) {
    union { float f; uint32_t u; } cv; cv.f = f;
    uint32_t u = cv.u;
    u += 0x7FFFu + ((u >> 16) & 1u);           // round-to-nearest-even
    return (unsigned short)(u >> 16);
}

// -----------------------------------------------------------------------------
// Kernel 1: one wave32 per row. Computes the wide scalar, builds bf16 deep_x
// (zero-padded K=224), and zero-inits the deep accumulator.
// -----------------------------------------------------------------------------
__global__ __launch_bounds__(256) void prep_kernel(
    const int*   __restrict__ sparse,   // [NB,3]
    const float* __restrict__ dense,    // [NB,13]
    const float* __restrict__ wide_w,
    const float* __restrict__ wide_b,
    const float* __restrict__ emb,      // [3,256,64]
    unsigned short* __restrict__ deep_x,// [NB,224] bf16
    float* __restrict__ wide_out,       // [NB]
    float* __restrict__ deep_acc)       // [NB]
{
    const int gid  = blockIdx.x * blockDim.x + threadIdx.x;
    const int row  = gid >> 5;
    const int lane = gid & 31;
    if (row >= NB) return;

    const int s0 = sparse[row * 3 + 0];
    const int s1 = sparse[row * 3 + 1];
    const int s2 = sparse[row * 3 + 2];

    // deep_x row: [emb0 | emb1 | emb2 | dense | zero-pad]
    for (int k = lane; k < DEEP_K; k += 32) {
        float v = 0.f;
        if (k < NF * ED) {
            const int f = k >> 6;                 // 64 per feature
            const int s = (f == 0) ? s0 : (f == 1 ? s1 : s2);
            v = emb[(f * NC + s) * ED + (k & 63)];
        } else if (k < DEEP_IN) {
            v = dense[row * NDN + (k - NF * ED)];
        }
        deep_x[(size_t)row * DEEP_K + k] = f2bf(v);
    }

    // wide: distribute the gathers across lanes, wave-reduce
    float w = 0.f;
    if (lane == 0) {
        w = wide_b[0] + wide_w[s0] + wide_w[NC + s1] + wide_w[2 * NC + s2];
    } else if (lane == 1) {
        w = wide_w[OFF_P01 + s0 * NF + s1];
    } else if (lane == 2) {
        w = wide_w[OFF_P02 + s0 * NF + s2];
    } else if (lane == 3) {
        w = wide_w[OFF_P12 + s1 * NF + s2];
    } else if (lane == 4) {
        w = wide_w[OFF_T012 + (s0 * NF + s1) * NF + s2];
    } else if (lane >= 5 && lane < 5 + NDN) {
        const int j = lane - 5;
        w = dense[row * NDN + j] * wide_w[OFF_DENSE + j];
    }
    for (int o = 16; o > 0; o >>= 1) w += __shfl_xor(w, o, 32);
    if (lane == 0) { wide_out[row] = w; deep_acc[row] = 0.f; }
}

// -----------------------------------------------------------------------------
// Weight conversion: fp32 -> bf16 (W1 zero-padded K 205->224)
// -----------------------------------------------------------------------------
__global__ void convW1(const float* __restrict__ W1, unsigned short* __restrict__ W1b) {
    const int i = blockIdx.x * blockDim.x + threadIdx.x;
    if (i >= NH * DEEP_K) return;
    const int n = i / DEEP_K, k = i - n * DEEP_K;
    W1b[i] = (k < DEEP_IN) ? f2bf(W1[n * DEEP_IN + k]) : (unsigned short)0;
}

__global__ void convW2(const float* __restrict__ W2, unsigned short* __restrict__ W2b) {
    const int i = blockIdx.x * blockDim.x + threadIdx.x;
    if (i < NH * NH) W2b[i] = f2bf(W2[i]);
}

// -----------------------------------------------------------------------------
// WMMA GEMM: out = relu(A @ Bw^T + bias).  A: [M,K] bf16 row-major,
// Bw: [N,K] bf16 row-major (so B[k,n] = Bw[n,k], column-major K x N).
// Per-wave tile: 32 rows x 64 cols  (2x4 of 16x16x32 bf16 WMMAs).
// Fragment layouts per CDNA5 ISA 7.12.2:
//   A lane: row = lane&15, holds K = kOff..kOff+7 and kOff+16..kOff+23,
//           kOff = (lane>>4)*8  -> two contiguous 16B chunks.
//   B lane: col = lane&15, same K as A (kOff..kOff+7, kOff+16..kOff+23).
//   C/D:    elem r -> row (lane>>4)*8 + r, col = lane&15.
// FUSED=true: epilogue multiplies by W3[n], reduces the 64-col strip per row
// and atomically accumulates into deep_acc[m] (h2 never hits memory).
// -----------------------------------------------------------------------------
template <bool FUSED>
__global__ __launch_bounds__(256) void gemm_wmma(
    const unsigned short* __restrict__ A,
    const unsigned short* __restrict__ Bw,
    const float* __restrict__ bias,
    int K,
    unsigned short* __restrict__ Cout,   // [M,NH] bf16 (if !FUSED)
    const float* __restrict__ W3,        // [NH]   (if FUSED)
    float* __restrict__ deep_acc)        // [NB]   (if FUSED)
{
    const int lane  = threadIdx.x & 31;
    const int wid   = threadIdx.x >> 5;
    const int mTile = blockIdx.x >> 1;               // 32-row tiles (M/32 = 512)
    const int nTile = (blockIdx.x & 1) * 8 + wid;    // 64-col tiles (N/64 = 16)
    const int M0 = mTile * 32;
    const int N0 = nTile * 64;

    v8f acc[2][4];
    const v8f vzero = {0.f, 0.f, 0.f, 0.f, 0.f, 0.f, 0.f, 0.f};
    #pragma unroll
    for (int mt = 0; mt < 2; ++mt)
        #pragma unroll
        for (int nt = 0; nt < 4; ++nt) acc[mt][nt] = vzero;

    const int aRow  = M0 + (lane & 15);
    const int aKoff = (lane >> 4) * 8;
    const int bCol  = N0 + (lane & 15);
    const int bKoff = (lane >> 4) * 8;

    for (int k0 = 0; k0 < K; k0 += 32) {
        Frag a[2], b[4];
        #pragma unroll
        for (int mt = 0; mt < 2; ++mt) {
            const unsigned short* p = A + (size_t)(aRow + mt * 16) * K + k0 + aKoff;
            a[mt].q[0] = *(const uint4*)(p);
            a[mt].q[1] = *(const uint4*)(p + 16);
        }
        #pragma unroll
        for (int nt = 0; nt < 4; ++nt) {
            const unsigned short* p = Bw + (size_t)(bCol + nt * 16) * K + k0 + bKoff;
            b[nt].q[0] = *(const uint4*)(p);
            b[nt].q[1] = *(const uint4*)(p + 16);
        }
        #pragma unroll
        for (int mt = 0; mt < 2; ++mt)
            #pragma unroll
            for (int nt = 0; nt < 4; ++nt)
                acc[mt][nt] = __builtin_amdgcn_wmma_f32_16x16x32_bf16(
                    false, a[mt].v, false, b[nt].v,
                    (short)0, acc[mt][nt], false, false);
    }

    const int nLocal = lane & 15;
    const int rBase  = (lane >> 4) * 8;

    if (!FUSED) {
        #pragma unroll
        for (int mt = 0; mt < 2; ++mt)
            #pragma unroll
            for (int nt = 0; nt < 4; ++nt) {
                const int n = N0 + nt * 16 + nLocal;
                const float bv = bias[n];
                #pragma unroll
                for (int r = 0; r < 8; ++r) {
                    float v = acc[mt][nt][r] + bv;
                    v = v > 0.f ? v : 0.f;
                    const int m = M0 + mt * 16 + rBase + r;
                    Cout[(size_t)m * NH + n] = f2bf(v);
                }
            }
    } else {
        #pragma unroll
        for (int mt = 0; mt < 2; ++mt) {
            float rp[8] = {0.f, 0.f, 0.f, 0.f, 0.f, 0.f, 0.f, 0.f};
            #pragma unroll
            for (int nt = 0; nt < 4; ++nt) {
                const int n = N0 + nt * 16 + nLocal;
                const float bv = bias[n];
                const float w3 = W3[n];
                #pragma unroll
                for (int r = 0; r < 8; ++r) {
                    float v = acc[mt][nt][r] + bv;
                    v = v > 0.f ? v : 0.f;
                    rp[r] += v * w3;
                }
            }
            #pragma unroll
            for (int r = 0; r < 8; ++r) {
                float v = rp[r];
                for (int o = 8; o >= 1; o >>= 1) v += __shfl_xor(v, o, 32);
                if (nLocal == 0)
                    atomicAdd(&deep_acc[M0 + mt * 16 + rBase + r], v);
            }
        }
    }
}

// -----------------------------------------------------------------------------
// Final: sigmoid(wide + deep + b3)
// -----------------------------------------------------------------------------
__global__ void final_kernel(const float* __restrict__ wide,
                             const float* __restrict__ deep,
                             const float* __restrict__ b3,
                             float* __restrict__ out)
{
    const int i = blockIdx.x * blockDim.x + threadIdx.x;
    if (i < NB) {
        const float x = wide[i] + deep[i] + b3[0];
        out[i] = 1.f / (1.f + __expf(-x));
    }
}

// -----------------------------------------------------------------------------
// Launch. Workspace layout (bytes, all 256-aligned; total ~41.6 MiB):
//   deep_x  bf16 [NB,224]    @ 0         (7,340,032)
//   W1b     bf16 [1024,224]  @ 7340032   (  458,752)
//   W2b     bf16 [1024,1024] @ 7798784   (2,097,152)
//   h1      bf16 [NB,1024]   @ 9895936   (33,554,432)
//   wide    f32  [NB]        @ 43450368  (   65,536)
//   deep    f32  [NB]        @ 43515904  (   65,536)
// -----------------------------------------------------------------------------
extern "C" void kernel_launch(void* const* d_in, const int* in_sizes, int n_in,
                              void* d_out, int out_size, void* d_ws, size_t ws_size,
                              hipStream_t stream) {
    const int*   sparse = (const int*)  d_in[0];
    const float* dense  = (const float*)d_in[1];
    const float* wide_w = (const float*)d_in[2];
    const float* wide_b = (const float*)d_in[3];
    const float* emb    = (const float*)d_in[4];
    const float* W1     = (const float*)d_in[5];
    const float* b1     = (const float*)d_in[6];
    const float* W2     = (const float*)d_in[7];
    const float* b2     = (const float*)d_in[8];
    const float* W3     = (const float*)d_in[9];
    const float* b3     = (const float*)d_in[10];
    float* out = (float*)d_out;

    char* ws = (char*)d_ws;
    unsigned short* deep_x = (unsigned short*)(ws);
    unsigned short* W1b    = (unsigned short*)(ws + 7340032);
    unsigned short* W2b    = (unsigned short*)(ws + 7798784);
    unsigned short* h1     = (unsigned short*)(ws + 9895936);
    float*          wide   = (float*)(ws + 43450368);
    float*          deep   = (float*)(ws + 43515904);

    prep_kernel<<<dim3((NB * 32) / 256), dim3(256), 0, stream>>>(
        sparse, dense, wide_w, wide_b, emb, deep_x, wide, deep);

    convW1<<<dim3((NH * DEEP_K + 255) / 256), dim3(256), 0, stream>>>(W1, W1b);
    convW2<<<dim3((NH * NH) / 256), dim3(256), 0, stream>>>(W2, W2b);

    // GEMM1: h1 = relu(deep_x @ W1b^T + b1)    M=16384, N=1024, K=224
    gemm_wmma<false><<<dim3(1024), dim3(256), 0, stream>>>(
        deep_x, W1b, b1, DEEP_K, h1, nullptr, nullptr);

    // GEMM2 (+ fused W3 matvec): deep += relu(h1 @ W2b^T + b2) @ W3
    gemm_wmma<true><<<dim3(1024), dim3(256), 0, stream>>>(
        h1, W2b, b2, NH, nullptr, W3, deep);

    final_kernel<<<dim3(NB / 256), dim3(256), 0, stream>>>(wide, deep, b3, out);
}
